// NSMLayer_7043746365773
// MI455X (gfx1250) — compile-verified, matrix operand order checked by reading, not run on hardware
//
#include <hip/hip_runtime.h>
#include <hip/hip_bf16.h>
#include <math.h>

// Problem constants (from reference)
#define B_    16
#define M_    2000
#define F_    200
#define H_    3
#define D_    64
#define HID_  768
#define R_    200
#define E_    160000
#define N_    (B_ * M_)     // 32000
#define HD_   (H_ * D_)     // 192
#define HD2_  (2 * HD_)     // 384
#define F3_   (3 * F_)      // 600

typedef __attribute__((ext_vector_type(16))) __bf16 v16bf;
typedef __attribute__((ext_vector_type(8)))  float  v8f;

// ---------------------------------------------------------------------------
// WMMA tile layouts (wave32, one wave per 16x16 f32 tile, K=32 bf16 per step)
// A 16x32: lane L -> row m=L&15, half=L>>4;
//   elems[0..7]  = A[m, kbase + half*8 + 0..7]
//   elems[8..15] = A[m, kbase + 16 + half*8 + 0..7]
// B 32x16 mirrored with n = L&15.  C/D v8f: elem v at (m=v+8*half, n=L&15).
// ---------------------------------------------------------------------------
__device__ __forceinline__ v8f wmma_bf16(v16bf a, v16bf b, v8f c) {
  return __builtin_amdgcn_wmma_f32_16x16x32_bf16(false, a, false, b, (short)0, c,
                                                 false, false);
}

__device__ __forceinline__ v16bf pack16(float4 a0, float4 a1, float4 b0, float4 b1) {
  v16bf v;
  v[0] = (__bf16)a0.x;  v[1] = (__bf16)a0.y;  v[2] = (__bf16)a0.z;  v[3] = (__bf16)a0.w;
  v[4] = (__bf16)a1.x;  v[5] = (__bf16)a1.y;  v[6] = (__bf16)a1.z;  v[7] = (__bf16)a1.w;
  v[8] = (__bf16)b0.x;  v[9] = (__bf16)b0.y;  v[10] = (__bf16)b0.z; v[11] = (__bf16)b0.w;
  v[12] = (__bf16)b1.x; v[13] = (__bf16)b1.y; v[14] = (__bf16)b1.z; v[15] = (__bf16)b1.w;
  return v;
}

// A-side loader: chunk-granular guards (every K here is a multiple of 8, so an
// 8-float chunk is entirely in or out of bounds -> 2 predicated b128-pair loads)
__device__ __forceinline__ v16bf load_a_dense(const float* __restrict__ A, int lda,
                                              int row0, int kbase, int kmax) {
  const int lane = threadIdx.x & 31;
  const int m = lane & 15, half = lane >> 4;
  const float* ar = A + (size_t)(row0 + m) * lda;
  const int k0 = kbase + half * 8;
  const int k1 = k0 + 16;
  float4 z = {0.0f, 0.0f, 0.0f, 0.0f};
  float4 a0 = z, a1 = z, b0 = z, b1 = z;
  if (k0 + 8 <= kmax) {
    const float4* p = (const float4*)(ar + k0);
    a0 = p[0]; a1 = p[1];
  }
  if (k1 + 8 <= kmax) {
    const float4* p = (const float4*)(ar + k1);
    b0 = p[0]; b1 = p[1];
  }
  return pack16(a0, a1, b0, b1);
}

// B-side: pre-packed bf16 fragments, one aligned 32B load per wave-lane
__device__ __forceinline__ v16bf load_b_packed(const __bf16* __restrict__ Bp,
                                               int kt, int ct, int ctiles) {
  const int lane = threadIdx.x & 31;
  return *(const v16bf*)(Bp + (((size_t)kt * ctiles + ct) * 32 + lane) * 16);
}

// ---------------------------------------------------------------------------
// Weight packing: W (K x C, row-major f32) -> [kt][ct][lane][16] bf16, padded
// ---------------------------------------------------------------------------
__global__ void k_pack_b(const float* __restrict__ W, int K, int C,
                         int ktiles, int ctiles, __bf16* __restrict__ out) {
  const int idx = blockIdx.x * blockDim.x + threadIdx.x;
  if (idx >= ktiles * ctiles * 32) return;
  const int lane = idx & 31;
  const int tile = idx >> 5;
  const int ct = tile % ctiles, kt = tile / ctiles;
  const int n = lane & 15, half = lane >> 4;
  const int col = ct * 16 + n;
  const int k0 = kt * 32 + half * 8;
  const int k1 = k0 + 16;
  __bf16* o = out + (size_t)idx * 16;
#pragma unroll
  for (int i = 0; i < 8; ++i) {
    float f0 = (col < C && k0 + i < K) ? W[(size_t)(k0 + i) * C + col] : 0.0f;
    float f1 = (col < C && k1 + i < K) ? W[(size_t)(k1 + i) * C + col] : 0.0f;
    o[i] = (__bf16)f0;
    o[i + 8] = (__bf16)f1;
  }
}

// ---------------------------------------------------------------------------
// Generic dense GEMM: out[row,col] = A[row,:] @ W + bias   (A is row-major,
// lda == KMAX for all uses).  grid (rows/16, ceil(NCOLS/16)), block 32.
// ---------------------------------------------------------------------------
template <int KMAX, int NCOLS>
__global__ void k_gemm_dense(const float* __restrict__ A,
                             const __bf16* __restrict__ Bp,
                             const float* __restrict__ bias,
                             float* __restrict__ out) {
  constexpr int KT = (KMAX + 31) / 32;
  constexpr int CT = (NCOLS + 15) / 16;
  const int rt = blockIdx.x, ct = blockIdx.y;
  const int row0 = rt * 16, col0 = ct * 16;
  __builtin_prefetch(A + (size_t)row0 * KMAX, 0, 1);
  v8f c = {};
  for (int kt = 0; kt < KT; ++kt) {
    v16bf a = load_a_dense(A, KMAX, row0, kt * 32, KMAX);
    v16bf b = load_b_packed(Bp, kt, ct, CT);
    c = wmma_bf16(a, b, c);
  }
  const int lane = threadIdx.x & 31;
  const int half = lane >> 4;
  const int col = col0 + (lane & 15);
  if (col < NCOLS) {
#pragma unroll
    for (int v = 0; v < 8; ++v) {
      const int row = row0 + v + half * 8;
      out[(size_t)row * NCOLS + col] = c[v] + bias[col];
    }
  }
}

// ---------------------------------------------------------------------------
// node_features = (nodes @ W_node + b_node) * instr[row/M]
// grid (N/16, 12), block 32
// ---------------------------------------------------------------------------
__global__ void k_gemm_nodeproj(const float* __restrict__ nodes,
                                const __bf16* __restrict__ Bp,
                                const float* __restrict__ bn,
                                const float* __restrict__ instr,
                                float* __restrict__ nfeat) {
  const int rt = blockIdx.x, ct = blockIdx.y;
  const int row0 = rt * 16, col0 = ct * 16;
  __builtin_prefetch(nodes + (size_t)row0 * F_, 0, 1);
  v8f c = {};
  for (int kt = 0; kt < 7; ++kt) {
    v16bf a = load_a_dense(nodes, F_, row0, kt * 32, F_);
    v16bf b = load_b_packed(Bp, kt, ct, HD_ / 16);
    c = wmma_bf16(a, b, c);
  }
  const int lane = threadIdx.x & 31;
  const int half = lane >> 4;
  const int col = col0 + (lane & 15);
#pragma unroll
  for (int v = 0; v < 8; ++v) {
    const int row = row0 + v + half * 8;
    const int bb = row / M_;
    nfeat[(size_t)row * HD_ + col] = (c[v] + bn[col]) * instr[bb * HD_ + col];
  }
}

// ---------------------------------------------------------------------------
// edge_features = (edges @ W_edge + b_edge) * instr[batch_ids[e]]
// grid (E/16, 12), block 32
// ---------------------------------------------------------------------------
__global__ void k_gemm_edgeproj(const float* __restrict__ edges,
                                const __bf16* __restrict__ Bp,
                                const float* __restrict__ be,
                                const float* __restrict__ instr,
                                const int* __restrict__ batch_ids,
                                float* __restrict__ efeat) {
  const int rt = blockIdx.x, ct = blockIdx.y;
  const int row0 = rt * 16, col0 = ct * 16;
  __builtin_prefetch(edges + (size_t)row0 * R_, 0, 1);
  v8f c = {};
  for (int kt = 0; kt < 7; ++kt) {
    v16bf a = load_a_dense(edges, R_, row0, kt * 32, R_);
    v16bf b = load_b_packed(Bp, kt, ct, HD_ / 16);
    c = wmma_bf16(a, b, c);
  }
  const int lane = threadIdx.x & 31;
  const int half = lane >> 4;
  const int col = col0 + (lane & 15);
#pragma unroll
  for (int v = 0; v < 8; ++v) {
    const int row = row0 + v + half * 8;
    const int bb = batch_ids[row];
    efeat[(size_t)row * HD_ + col] = (c[v] + be[col]) * instr[bb * HD_ + col];
  }
}

// ---------------------------------------------------------------------------
// Gathered NL GEMM + relu + attention scale + atomic scatter.
//   rows r = e*H + h ; A[r,k] = k<64 ? nfeat[gidx[e],h,k] : efeat[e,h,k-64]
//   acc[sidx[e],h,col] += attn(e,h) * relu(A @ W_nl + b_nl)
// Chunks (8 floats, multiples of 8) never straddle the k=64 concat boundary,
// so the gather is a branch-free base-pointer select + aligned float4 loads.
// grid (E*H/16, 4), block 32
// ---------------------------------------------------------------------------
__global__ void k_gemm_nl_scatter(const float* __restrict__ nfeat,
                                  const float* __restrict__ efeat,
                                  const int* __restrict__ gidx,
                                  const int* __restrict__ sidx,
                                  const __bf16* __restrict__ Bp,
                                  const float* __restrict__ bnl,
                                  const float* __restrict__ expS,
                                  const float* __restrict__ den,
                                  const float* __restrict__ prior,
                                  float* __restrict__ acc) {
  const int rt = blockIdx.x, ct = blockIdx.y;
  const int row0 = rt * 16, col0 = ct * 16;
  const int lane = threadIdx.x & 31;
  const int mrow = lane & 15, half = lane >> 4;
  const int r = row0 + mrow;
  const int e = r / H_, h = r - e * H_;
  const float* nrow = nfeat + (size_t)gidx[e] * HD_ + h * D_;
  const float* erow = efeat + (size_t)e * HD_ + h * D_;
  v8f c = {};
#pragma unroll
  for (int kt = 0; kt < 4; ++kt) {
    const int k0 = kt * 32 + half * 8;
    const int k1 = k0 + 16;
    const float* p0 = (k0 < D_) ? (nrow + k0) : (erow + (k0 - D_));
    const float* p1 = (k1 < D_) ? (nrow + k1) : (erow + (k1 - D_));
    float4 a0 = ((const float4*)p0)[0], a1 = ((const float4*)p0)[1];
    float4 b0 = ((const float4*)p1)[0], b1 = ((const float4*)p1)[1];
    v16bf a = pack16(a0, a1, b0, b1);
    v16bf b = load_b_packed(Bp, kt, ct, D_ / 16);
    c = wmma_bf16(a, b, c);
  }
  const int col = col0 + (lane & 15);  // in [0,64)
#pragma unroll
  for (int v = 0; v < 8; ++v) {
    const int rr = row0 + v + half * 8;
    const int ee = rr / H_, hh = rr - ee * H_;
    float g = c[v] + bnl[col];
    g = g > 0.0f ? g : 0.0f;
    const int gn = gidx[ee];
    const float es = expS[ee * H_ + hh];
    const float w = (es / (den[gn * H_ + hh] + 1e-16f)) * prior[gn * H_ + hh];
    atomicAdd(&acc[(size_t)sidx[ee] * HD_ + hh * D_ + col], w * g);
  }
}

// ---------------------------------------------------------------------------
// Small elementwise / reduction kernels
// ---------------------------------------------------------------------------
__global__ void k_zero(float* __restrict__ p, size_t n) {
  size_t i = (size_t)blockIdx.x * blockDim.x + threadIdx.x;
  const size_t stride = (size_t)gridDim.x * blockDim.x;
  for (; i < n; i += stride) p[i] = 0.0f;
}

__global__ void k_nscore(const float* __restrict__ nfeat,
                         const float* __restrict__ psf,
                         const float* __restrict__ act,
                         float* __restrict__ scores) {
  const int i = blockIdx.x * blockDim.x + threadIdx.x;
  if (i >= N_ * H_) return;
  const int n = i / H_, h = i - n * H_;
  const float* nf = nfeat + (size_t)n * HD_ + h * D_;
  const float* ps = psf + h * D_;
  float s = 0.0f;
  for (int d = 0; d < D_; ++d) {
    const float x = nf[d] * ps[d];
    s += x > 0.0f ? x : 0.2f * x;   // leaky_relu(0.2)
  }
  const float a = act[n];
  scores[i] = s * a + (1.0f - a) * -1e20f;
}

// softmax over M per (b,h); scores buffer becomes prior, in place
__global__ void k_softmax_m(float* __restrict__ scores) {
  __shared__ float red[256];
  const int bh = blockIdx.x;
  const int b = bh / H_, h = bh - b * H_;
  const int tid = threadIdx.x;
  float lm = -INFINITY;
  for (int m = tid; m < M_; m += 256)
    lm = fmaxf(lm, scores[(size_t)(b * M_ + m) * H_ + h]);
  red[tid] = lm;
  __syncthreads();
  for (int s = 128; s > 0; s >>= 1) {
    if (tid < s) red[tid] = fmaxf(red[tid], red[tid + s]);
    __syncthreads();
  }
  const float mx = red[0];
  __syncthreads();
  float ls = 0.0f;
  for (int m = tid; m < M_; m += 256)
    ls += expf(scores[(size_t)(b * M_ + m) * H_ + h] - mx);
  red[tid] = ls;
  __syncthreads();
  for (int s = 128; s > 0; s >>= 1) {
    if (tid < s) red[tid] += red[tid + s];
    __syncthreads();
  }
  const float inv = 1.0f / red[0];
  for (int m = tid; m < M_; m += 256) {
    const size_t idx = (size_t)(b * M_ + m) * H_ + h;
    scores[idx] = expf(scores[idx] - mx) * inv;
  }
}

// edge attention numerators + segment denominators (global-max shift cancels
// in exp_s/denom ratio, so it is omitted)
__global__ void k_escore(const float* __restrict__ efeat,
                         const float* __restrict__ esf,
                         const int* __restrict__ src, const int* __restrict__ trg,
                         float* __restrict__ expS,
                         float* __restrict__ denS, float* __restrict__ denT) {
  const int i = blockIdx.x * blockDim.x + threadIdx.x;
  if (i >= E_ * H_) return;
  const int e = i / H_, h = i - e * H_;
  const float* ef = efeat + (size_t)e * HD_ + h * D_;
  const float* es = esf + h * D_;
  float s = 0.0f;
  for (int d = 0; d < D_; ++d) {
    const float x = ef[d] * es[d];
    s += x > 0.0f ? x : 0.2f * x;
  }
  const float v = expf(s);
  expS[i] = v;
  atomicAdd(&denS[src[e] * H_ + h], v);
  atomicAdd(&denT[trg[e] * H_ + h], v);
}

__global__ void k_deg(const int* __restrict__ src, const int* __restrict__ trg,
                      const float* __restrict__ act, float* __restrict__ deg) {
  const int e = blockIdx.x * blockDim.x + threadIdx.x;
  if (e >= E_) return;
  atomicAdd(&deg[trg[e]], act[src[e]]);
  atomicAdd(&deg[src[e]], act[trg[e]]);
}

// LayerNorm over concat([inward, outward] per head) -> newf (N x 384)
__global__ void k_lnagg(const float* __restrict__ inew, const float* __restrict__ onew,
                        const float* __restrict__ gamma, const float* __restrict__ beta,
                        float* __restrict__ newf) {
  __shared__ float red[512];
  __shared__ float red2[512];
  const int n = blockIdx.x;
  const int tid = threadIdx.x;  // 512 threads, 384 active
  float x = 0.0f;
  if (tid < HD2_) {
    const int h = tid / (2 * D_), r = tid - h * 2 * D_;
    x = (r < D_) ? inew[(size_t)n * HD_ + h * D_ + r]
                 : onew[(size_t)n * HD_ + h * D_ + (r - D_)];
  }
  red[tid]  = (tid < HD2_) ? x : 0.0f;
  red2[tid] = (tid < HD2_) ? x * x : 0.0f;
  __syncthreads();
  for (int s = 256; s > 0; s >>= 1) {
    if (tid < s) { red[tid] += red[tid + s]; red2[tid] += red2[tid + s]; }
    __syncthreads();
  }
  const float mean = red[0] / (float)HD2_;
  const float var = red2[0] / (float)HD2_ - mean * mean;
  const float inv = rsqrtf(var + 1e-5f);
  if (tid < HD2_)
    newf[(size_t)n * HD2_ + tid] = (x - mean) * inv * gamma[tid] + beta[tid];
}

// GRU gates + residual LayerNorm(F) -> out
__global__ void k_gru(const float* __restrict__ ih, const float* __restrict__ hh,
                      const float* __restrict__ nodes,
                      const float* __restrict__ g, const float* __restrict__ bta,
                      float* __restrict__ out) {
  __shared__ float red[256];
  __shared__ float red2[256];
  const int n = blockIdx.x;
  const int tid = threadIdx.x;  // 256 threads, 200 active
  float x = 0.0f;
  if (tid < F_) {
    const float* ihp = ih + (size_t)n * F3_;
    const float* hhp = hh + (size_t)n * F3_;
    const float i0 = ihp[tid],          h0 = hhp[tid];
    const float i1 = ihp[tid + F_],     h1 = hhp[tid + F_];
    const float i2 = ihp[tid + 2 * F_], h2 = hhp[tid + 2 * F_];
    const float nd = nodes[(size_t)n * F_ + tid];
    const float upd = 1.0f / (1.0f + expf(-(i0 + h0)));
    const float rst = 1.0f / (1.0f + expf(-(i1 + h1)));
    const float mem = tanhf(i2 + rst * h2);
    const float o = (1.0f - upd) * mem + upd * nd;
    x = o + nd;  // residual
  }
  red[tid]  = (tid < F_) ? x : 0.0f;
  red2[tid] = (tid < F_) ? x * x : 0.0f;
  __syncthreads();
  for (int s = 128; s > 0; s >>= 1) {
    if (tid < s) { red[tid] += red[tid + s]; red2[tid] += red2[tid + s]; }
    __syncthreads();
  }
  const float mean = red[0] / (float)F_;
  const float var = red2[0] / (float)F_ - mean * mean;
  const float inv = rsqrtf(var + 1e-5f);
  if (tid < F_)
    out[(size_t)n * F_ + tid] = (x - mean) * inv * g[tid] + bta[tid];
}

__global__ void k_nextact(const float* __restrict__ deg, float* __restrict__ na) {
  const int n = blockIdx.x * blockDim.x + threadIdx.x;
  if (n < N_) na[n] = deg[n] > 0.0f ? 1.0f : 0.0f;
}

// ---------------------------------------------------------------------------
extern "C" void kernel_launch(void* const* d_in, const int* in_sizes, int n_in,
                              void* d_out, int out_size, void* d_ws, size_t ws_size,
                              hipStream_t stream) {
  const float* nodes       = (const float*)d_in[0];
  const float* edges       = (const float*)d_in[1];
  const float* instruction = (const float*)d_in[2];
  const int*   edge_index  = (const int*)d_in[3];
  const int*   batch_ids   = (const int*)d_in[4];
  const float* activation  = (const float*)d_in[5];
  // d_in[6] node_mask: unused by the reference math
  const float* W_node  = (const float*)d_in[7];
  const float* b_node  = (const float*)d_in[8];
  const float* W_instr = (const float*)d_in[9];
  const float* b_instr = (const float*)d_in[10];
  const float* W_edge  = (const float*)d_in[11];
  const float* b_edge  = (const float*)d_in[12];
  const float* psf     = (const float*)d_in[13];
  const float* esf     = (const float*)d_in[14];
  const float* W_nl    = (const float*)d_in[15];
  const float* b_nl    = (const float*)d_in[16];
  const float* W_ih    = (const float*)d_in[17];
  const float* b_ih    = (const float*)d_in[18];
  const float* W_hh    = (const float*)d_in[19];
  const float* b_hh    = (const float*)d_in[20];
  const float* agg_g   = (const float*)d_in[21];
  const float* agg_b   = (const float*)d_in[22];
  const float* ln_g    = (const float*)d_in[23];
  const float* ln_b    = (const float*)d_in[24];
  const int* src = edge_index;
  const int* trg = edge_index + E_;

  // --- packed-bf16 weight pool at start of workspace (32B-aligned base) ---
  // sizes in v16bf fragments: ktiles * ctiles * 32 lanes * 16 elems
  const int KT_NODE = 7,  CT_HD = HD_ / 16;          // 7 x 12
  const int KT_INSTR = HID_ / 32;                    // 24
  const int KT_NL = 4,  CT_NL = D_ / 16;             // 4 x 4
  const int KT_IH = HD2_ / 32;                       // 12
  const int KT_HH = 7;
  const int CT_F3 = (F3_ + 15) / 16;                 // 38
  __bf16* bp = (__bf16*)d_ws;
  __bf16* pB_node  = bp;  bp += (size_t)KT_NODE  * CT_HD * 512;
  __bf16* pB_edge  = bp;  bp += (size_t)KT_NODE  * CT_HD * 512;
  __bf16* pB_instr = bp;  bp += (size_t)KT_INSTR * CT_HD * 512;
  __bf16* pB_nl    = bp;  bp += (size_t)KT_NL    * CT_NL * 512;
  __bf16* pB_ih    = bp;  bp += (size_t)KT_IH    * CT_F3 * 512;
  __bf16* pB_hh    = bp;  bp += (size_t)KT_HH    * CT_F3 * 512;

  // --- float workspace after the packed pool ---
  float* w = (float*)bp;
  float* instr = w;  w += (size_t)B_ * HD_;
  float* nfeat = w;  w += (size_t)N_ * HD_;
  float* prior = w;  w += (size_t)N_ * H_;    // scores -> prior (in place)
  float* efeat = w;  w += (size_t)E_ * HD_;
  float* expS  = w;  w += (size_t)E_ * H_;
  float* denS  = w;  w += (size_t)N_ * H_;    // |-- zeroed region start
  float* denT  = w;  w += (size_t)N_ * H_;
  float* deg   = w;  w += (size_t)N_;
  float* onew  = w;  w += (size_t)N_ * HD_;
  float* inew  = w;  w += (size_t)N_ * HD_;   // |-- zeroed region end
  float* newf  = w;  w += (size_t)N_ * HD2_;
  float* ihb   = w;  w += (size_t)N_ * F3_;
  float* hhb   = w;  w += (size_t)N_ * F3_;

  const size_t nzero = (size_t)N_ * H_ * 2 + (size_t)N_ + (size_t)N_ * HD_ * 2;

  // 0) zero accumulators, pack all weights to WMMA-B bf16 fragments
  k_zero<<<2048, 256, 0, stream>>>(denS, nzero);
  auto packn = [](int kt, int ct) { return kt * ct * 32; };
  k_pack_b<<<(packn(KT_NODE, CT_HD) + 255) / 256, 256, 0, stream>>>(
      W_node, F_, HD_, KT_NODE, CT_HD, pB_node);
  k_pack_b<<<(packn(KT_NODE, CT_HD) + 255) / 256, 256, 0, stream>>>(
      W_edge, R_, HD_, KT_NODE, CT_HD, pB_edge);
  k_pack_b<<<(packn(KT_INSTR, CT_HD) + 255) / 256, 256, 0, stream>>>(
      W_instr, HID_, HD_, KT_INSTR, CT_HD, pB_instr);
  k_pack_b<<<(packn(KT_NL, CT_NL) + 255) / 256, 256, 0, stream>>>(
      W_nl, 2 * D_, D_, KT_NL, CT_NL, pB_nl);
  k_pack_b<<<(packn(KT_IH, CT_F3) + 255) / 256, 256, 0, stream>>>(
      W_ih, HD2_, F3_, KT_IH, CT_F3, pB_ih);
  k_pack_b<<<(packn(KT_HH, CT_F3) + 255) / 256, 256, 0, stream>>>(
      W_hh, F_, F3_, KT_HH, CT_F3, pB_hh);

  // 1) instruction projection via WMMA (16 rows -> 1 row tile)
  {
    dim3 grid(1, CT_HD);
    k_gemm_dense<HID_, HD_><<<grid, 32, 0, stream>>>(instruction, pB_instr,
                                                     b_instr, instr);
  }
  // 2) node projection GEMM fused with instr scaling
  {
    dim3 grid(N_ / 16, CT_HD);
    k_gemm_nodeproj<<<grid, 32, 0, stream>>>(nodes, pB_node, b_node, instr, nfeat);
  }
  // 3) node prior scores + softmax over M
  k_nscore<<<(N_ * H_ + 255) / 256, 256, 0, stream>>>(nfeat, psf, activation, prior);
  k_softmax_m<<<B_ * H_, 256, 0, stream>>>(prior);
  // 4) edge projection GEMM fused with instr scaling
  {
    dim3 grid(E_ / 16, CT_HD);
    k_gemm_edgeproj<<<grid, 32, 0, stream>>>(edges, pB_edge, b_edge, instr,
                                             batch_ids, efeat);
  }
  // 5) edge attention numerators + segment denominators; activation degrees
  k_escore<<<(E_ * H_ + 255) / 256, 256, 0, stream>>>(efeat, esf, src, trg,
                                                      expS, denS, denT);
  k_deg<<<(E_ + 255) / 256, 256, 0, stream>>>(src, trg, activation, deg);
  // 6) gathered NL GEMMs with fused relu + attention + atomic scatter
  {
    dim3 grid((E_ * H_) / 16, CT_NL);
    k_gemm_nl_scatter<<<grid, 32, 0, stream>>>(nfeat, efeat, src, trg, pB_nl,
                                               b_nl, expS, denS, prior, onew);
    k_gemm_nl_scatter<<<grid, 32, 0, stream>>>(nfeat, efeat, trg, src, pB_nl,
                                               b_nl, expS, denT, prior, inew);
  }
  // 7) LayerNorm over aggregated 384-dim features
  k_lnagg<<<N_, 512, 0, stream>>>(inew, onew, agg_g, agg_b, newf);
  // 8) GRU gate GEMMs
  {
    dim3 grid(N_ / 16, CT_F3);
    k_gemm_dense<HD2_, F3_><<<grid, 32, 0, stream>>>(newf, pB_ih, b_ih, ihb);
    k_gemm_dense<F_, F3_><<<grid, 32, 0, stream>>>(nodes, pB_hh, b_hh, hhb);
  }
  // 9) GRU gates + residual LayerNorm -> out ; next activation
  float* out_main = (float*)d_out;
  float* out_act  = out_main + (size_t)N_ * F_;
  k_gru<<<N_, 256, 0, stream>>>(ihb, hhb, nodes, ln_g, ln_b, out_main);
  k_nextact<<<(N_ + 255) / 256, 256, 0, stream>>>(deg, out_act);
}
